// EfficientGraphAttention_42631845380866
// MI455X (gfx1250) — matrix operand order repaired and checked by reading, not compiled, not run on hardware
//
#include <hip/hip_runtime.h>
#include <hip/hip_bf16.h>
#include <stdint.h>

// ---------------------------------------------------------------------------
// Types for CDNA5 WMMA (wave32): D(16x16 f32) = A(16x32 bf16) x B(32x16 bf16) + C
// ---------------------------------------------------------------------------
typedef __attribute__((ext_vector_type(16))) __bf16          v16bf;
typedef __attribute__((ext_vector_type(8)))  float           v8f;

#define DIM   128
#define HEADS 4
#define DH    32
#define NEG_SLOPE 0.2f
#define LN_EPS 1e-5f

// float -> bf16 round-to-nearest-even (bit-level)
__device__ __forceinline__ unsigned short f2bf(float f) {
    unsigned int u = __float_as_uint(f);
    unsigned int r = u + 0x7FFFu + ((u >> 16) & 1u);
    return (unsigned short)(r >> 16);
}

__device__ __forceinline__ v8f wmma_bf16(v16bf a, v16bf b, v8f c) {
    return __builtin_amdgcn_wmma_f32_16x16x32_bf16(
        /*neg_a=*/false, a, /*neg_b=*/false, b,
        /*c_mod=*/(short)0, c, /*reuse_a=*/false, /*reuse_b=*/false);
}

union Frag128x2 { uint4 q[2]; v16bf b; };

// A fragment (16x32 bf16) from bf16 row-major: two 16-byte loads.
// row = row0+(lane&15), kbase = kstep*32 + 8*(lane>>4) (multiple of 8 -> 16B aligned)
__device__ __forceinline__ v16bf load_a_bf16(const unsigned short* __restrict__ A,
                                             int lda, int row, int kbase) {
    const unsigned short* p = A + (size_t)row * lda + kbase;
    Frag128x2 t;
    t.q[0] = *(const uint4*)(p);
    t.q[1] = *(const uint4*)(p + 16);
    return t.b;
}

// B fragment from pre-packed layout [ntile][kstep][lane][16]: two 16-byte loads.
__device__ __forceinline__ v16bf load_b_packed(const unsigned short* __restrict__ Bp,
                                               int ntile, int kstep, int ksteps, int lane) {
    const unsigned short* p = Bp + ((size_t)(ntile * ksteps + kstep) * 32 + lane) * 16;
    Frag128x2 t;
    t.q[0] = *(const uint4*)(p);
    t.q[1] = *(const uint4*)(p + 8);
    return t.b;
}

// Float atomics on global memory (lower to global_atomic_add_f32 etc.)
__device__ __forceinline__ void atomic_add_f32(float* p, float v) {
    __hip_atomic_fetch_add(p, v, __ATOMIC_RELAXED, __HIP_MEMORY_SCOPE_AGENT);
}
__device__ __forceinline__ void atomic_max_f32(float* addr, float val) {
    if (val >= 0.f) atomicMax((int*)addr, __float_as_int(val));
    else            atomicMin((unsigned int*)addr, __float_as_uint(val));
}

// ---------------------------------------------------------------------------
// 0a) pack a f32 row-major weight [K x Ncols] into per-lane B-fragment layout
// ---------------------------------------------------------------------------
__global__ void pack_b_kernel(const float* __restrict__ W, int ldb,
                              int ntiles, int ksteps, unsigned short* __restrict__ out) {
    int idx = blockIdx.x * blockDim.x + threadIdx.x;
    int total = ntiles * ksteps * 32;
    if (idx >= total) return;
    const int lane  = idx & 31;
    const int kstep = (idx >> 5) % ksteps;
    const int ntile = idx / (32 * ksteps);
    const int col = ntile * 16 + (lane & 15);
    const int kb  = kstep * 32 + (lane >> 4) * 8;
    unsigned short* o = out + (size_t)idx * 16;
    for (int j = 0; j < 8; ++j) o[j]     = f2bf(W[(size_t)(kb + j)      * ldb + col]);
    for (int j = 0; j < 8; ++j) o[8 + j] = f2bf(W[(size_t)(kb + 16 + j) * ldb + col]);
}

// 0b) f32 -> bf16 row-major
__global__ void cvt_bf16_kernel(const float* __restrict__ in,
                                unsigned short* __restrict__ out, int n) {
    int i = blockIdx.x * blockDim.x + threadIdx.x;
    if (i < n) out[i] = f2bf(in[i]);
}

// 0c) init: accum = 0, seg-max = -inf, seg-sum = 0 (ws is not re-poisoned
//     between replays -> must run every call)
__global__ void init_kernel(float* __restrict__ m, float* __restrict__ z,
                            float* __restrict__ accum, int Nn) {
    int i = blockIdx.x * blockDim.x + threadIdx.x;
    if (i < Nn * DIM) accum[i] = 0.f;
    if (i < Nn * HEADS) { m[i] = -3.402823466e38f; z[i] = 0.f; }
}

// ---------------------------------------------------------------------------
// 1) h = x @ W_gat  (Npad x 128)(128 x 128); wave -> 16-row strip, 8 col tiles.
//    All buffers padded to the grid's row coverage -> no bounds checks at all.
// ---------------------------------------------------------------------------
__global__ void __launch_bounds__(256)
gemm_h_kernel(const unsigned short* __restrict__ xb,
              const unsigned short* __restrict__ Wp,
              float* __restrict__ h) {
    const int lane = threadIdx.x & 31;
    const int wave = threadIdx.x >> 5;
    const int row0 = (blockIdx.x * 8 + wave) * 16;
    const int arow = row0 + (lane & 15);
    const int kh = (lane >> 4) * 8;

    v16bf a0 = load_a_bf16(xb, DIM, arow,  0 + kh);
    v16bf a1 = load_a_bf16(xb, DIM, arow, 32 + kh);
    v16bf a2 = load_a_bf16(xb, DIM, arow, 64 + kh);
    v16bf a3 = load_a_bf16(xb, DIM, arow, 96 + kh);

    const int rbase = row0 + (lane >> 4) * 8;
#pragma unroll
    for (int nt = 0; nt < 8; ++nt) {
        __builtin_prefetch(Wp + (size_t)((nt + 1) * 4) * 32 * 16, 0, 0);
        v8f acc = {};
        acc = wmma_bf16(a0, load_b_packed(Wp, nt, 0, 4, lane), acc);
        acc = wmma_bf16(a1, load_b_packed(Wp, nt, 1, 4, lane), acc);
        acc = wmma_bf16(a2, load_b_packed(Wp, nt, 2, 4, lane), acc);
        acc = wmma_bf16(a3, load_b_packed(Wp, nt, 3, 4, lane), acc);
        float* p = h + (size_t)rbase * DIM + nt * 16 + (lane & 15);
#pragma unroll
        for (int j = 0; j < 8; ++j) p[(size_t)j * DIM] = acc[j];
    }
}

// ---------------------------------------------------------------------------
// 2) attention coefficients: a_s[n,h] = <h[n,h,:], att_src[h,:]>, same a_d
// ---------------------------------------------------------------------------
__global__ void attn_coef_kernel(const float* __restrict__ h,
                                 const float* __restrict__ att_src,
                                 const float* __restrict__ att_dst,
                                 float* __restrict__ a_s, float* __restrict__ a_d,
                                 int Nn) {
    int idx = blockIdx.x * blockDim.x + threadIdx.x;
    if (idx >= Nn * HEADS) return;
    const int node = idx >> 2, hd = idx & 3;
    const float* hp = h + (size_t)node * DIM + hd * DH;
    float s = 0.f, d = 0.f;
    for (int j = 0; j < DH; ++j) {
        float v = hp[j];
        s += v * att_src[hd * DH + j];
        d += v * att_dst[hd * DH + j];
    }
    a_s[idx] = s;
    a_d[idx] = d;
}

__device__ __forceinline__ void edge_ids(const long long* __restrict__ ei,
                                         int e, int E_, int& src, int& dst) {
    if (e < E_) { src = (int)ei[e]; dst = (int)ei[E_ + e]; }
    else        { src = dst = e - E_; }           // self-loop
}

// ---------------------------------------------------------------------------
// 3) segment max of leaky_relu(a_s[src]+a_d[dst]) over incoming edges
// ---------------------------------------------------------------------------
__global__ void edge_max_kernel(const long long* __restrict__ ei,
                                const float* __restrict__ a_s,
                                const float* __restrict__ a_d,
                                float* __restrict__ m, int E_, int Nn) {
    int e = blockIdx.x * blockDim.x + threadIdx.x;
    if (e >= E_ + Nn) return;
    int src, dst; edge_ids(ei, e, E_, src, dst);
    for (int hd = 0; hd < HEADS; ++hd) {
        float l = a_s[src * HEADS + hd] + a_d[dst * HEADS + hd];
        l = l > 0.f ? l : NEG_SLOPE * l;
        atomic_max_f32(&m[dst * HEADS + hd], l);
    }
}

// ---------------------------------------------------------------------------
// 4) segment sum of exp(logit - max)
// ---------------------------------------------------------------------------
__global__ void edge_sum_kernel(const long long* __restrict__ ei,
                                const float* __restrict__ a_s,
                                const float* __restrict__ a_d,
                                const float* __restrict__ m,
                                float* __restrict__ z, int E_, int Nn) {
    int e = blockIdx.x * blockDim.x + threadIdx.x;
    if (e >= E_ + Nn) return;
    int src, dst; edge_ids(ei, e, E_, src, dst);
    for (int hd = 0; hd < HEADS; ++hd) {
        float l = a_s[src * HEADS + hd] + a_d[dst * HEADS + hd];
        l = l > 0.f ? l : NEG_SLOPE * l;
        atomic_add_f32(&z[dst * HEADS + hd], __expf(l - m[dst * HEADS + hd]));
    }
}

// ---------------------------------------------------------------------------
// 5) message scatter: accum[dst, head, :] += alpha * h[src, head, :]
// ---------------------------------------------------------------------------
__global__ void edge_scatter_kernel(const long long* __restrict__ ei,
                                    const float* __restrict__ a_s,
                                    const float* __restrict__ a_d,
                                    const float* __restrict__ m,
                                    const float* __restrict__ z,
                                    const float* __restrict__ h,
                                    float* __restrict__ accum, int E_, int Nn) {
    int t = blockIdx.x * blockDim.x + threadIdx.x;
    if (t >= (E_ + Nn) * HEADS) return;
    const int e = t >> 2, hd = t & 3;
    int src, dst; edge_ids(ei, e, E_, src, dst);
    float l = a_s[src * HEADS + hd] + a_d[dst * HEADS + hd];
    l = l > 0.f ? l : NEG_SLOPE * l;
    float alpha = __expf(l - m[dst * HEADS + hd]) / (z[dst * HEADS + hd] + 1e-16f);
    const float* hs = h + (size_t)src * DIM + hd * DH;
    float*       ob = accum + (size_t)dst * DIM + hd * DH;
    for (int j = 0; j < DH; ++j) atomic_add_f32(&ob[j], hs[j] * alpha);
}

// ---------------------------------------------------------------------------
// 6) x1 = LayerNorm(x + accum + bias_gat); also emit bf16 copy for FFN GEMM
// ---------------------------------------------------------------------------
__global__ void __launch_bounds__(DIM)
ln1_kernel(const float* __restrict__ x, const float* __restrict__ accum,
           const float* __restrict__ bias, const float* __restrict__ lw,
           const float* __restrict__ lb, float* __restrict__ x1,
           unsigned short* __restrict__ x1b) {
    const int row = blockIdx.x, t = threadIdx.x;
    __shared__ float red[DIM];
    const size_t idx = (size_t)row * DIM + t;
    float v = x[idx] + accum[idx] + bias[t];
    red[t] = v; __syncthreads();
    for (int s = DIM / 2; s > 0; s >>= 1) { if (t < s) red[t] += red[t + s]; __syncthreads(); }
    float mu = red[0] * (1.f / DIM);
    __syncthreads();
    float d = v - mu;
    red[t] = d * d; __syncthreads();
    for (int s = DIM / 2; s > 0; s >>= 1) { if (t < s) red[t] += red[t + s]; __syncthreads(); }
    float o = d * rsqrtf(red[0] * (1.f / DIM) + LN_EPS) * lw[t] + lb[t];
    x1[idx]  = o;
    x1b[idx] = f2bf(o);
}

// ---------------------------------------------------------------------------
// 7) y = relu(x1 @ w1 + b1), (Npad x 256), output bf16 for GEMM2's A operand
// ---------------------------------------------------------------------------
__global__ void __launch_bounds__(256)
gemm_ffn1_kernel(const unsigned short* __restrict__ x1b,
                 const unsigned short* __restrict__ w1p,
                 const float* __restrict__ b1, unsigned short* __restrict__ y) {
    const int lane = threadIdx.x & 31;
    const int wave = threadIdx.x >> 5;
    const int row0 = (blockIdx.x * 8 + wave) * 16;
    const int arow = row0 + (lane & 15);
    const int kh = (lane >> 4) * 8;

    v16bf a0 = load_a_bf16(x1b, DIM, arow,  0 + kh);
    v16bf a1 = load_a_bf16(x1b, DIM, arow, 32 + kh);
    v16bf a2 = load_a_bf16(x1b, DIM, arow, 64 + kh);
    v16bf a3 = load_a_bf16(x1b, DIM, arow, 96 + kh);

    const int rbase = row0 + (lane >> 4) * 8;
#pragma unroll
    for (int nt = 0; nt < 16; ++nt) {
        v8f acc = {};
        acc = wmma_bf16(a0, load_b_packed(w1p, nt, 0, 4, lane), acc);
        acc = wmma_bf16(a1, load_b_packed(w1p, nt, 1, 4, lane), acc);
        acc = wmma_bf16(a2, load_b_packed(w1p, nt, 2, 4, lane), acc);
        acc = wmma_bf16(a3, load_b_packed(w1p, nt, 3, 4, lane), acc);
        const int c = nt * 16 + (lane & 15);
        const float bc = b1[c];
        unsigned short* p = y + (size_t)rbase * (2 * DIM) + c;
#pragma unroll
        for (int j = 0; j < 8; ++j) {
            float v = acc[j] + bc;
            p[(size_t)j * (2 * DIM)] = f2bf(v > 0.f ? v : 0.f);
        }
    }
}

// ---------------------------------------------------------------------------
// 8) f = y @ w2 + b2, (Npad x 128), K = 256 (8 WMMA steps)
// ---------------------------------------------------------------------------
__global__ void __launch_bounds__(256)
gemm_ffn2_kernel(const unsigned short* __restrict__ y,
                 const unsigned short* __restrict__ w2p,
                 const float* __restrict__ b2, float* __restrict__ f) {
    const int lane = threadIdx.x & 31;
    const int wave = threadIdx.x >> 5;
    const int row0 = (blockIdx.x * 8 + wave) * 16;
    const int arow = row0 + (lane & 15);
    const int kh = (lane >> 4) * 8;

    v16bf a[8];
#pragma unroll
    for (int ks = 0; ks < 8; ++ks)
        a[ks] = load_a_bf16(y, 2 * DIM, arow, ks * 32 + kh);

    const int rbase = row0 + (lane >> 4) * 8;
#pragma unroll
    for (int nt = 0; nt < 8; ++nt) {
        v8f acc = {};
#pragma unroll
        for (int ks = 0; ks < 8; ++ks)
            acc = wmma_bf16(a[ks], load_b_packed(w2p, nt, ks, 8, lane), acc);
        const int c = nt * 16 + (lane & 15);
        const float bc = b2[c];
        float* p = f + (size_t)rbase * DIM + c;
#pragma unroll
        for (int j = 0; j < 8; ++j) p[(size_t)j * DIM] = acc[j] + bc;
    }
}

// ---------------------------------------------------------------------------
// 9) out = LayerNorm(x1 + f)
// ---------------------------------------------------------------------------
__global__ void __launch_bounds__(DIM)
ln2_kernel(const float* __restrict__ x1, const float* __restrict__ f,
           const float* __restrict__ lw, const float* __restrict__ lb,
           float* __restrict__ out) {
    const int row = blockIdx.x, t = threadIdx.x;
    __shared__ float red[DIM];
    const size_t idx = (size_t)row * DIM + t;
    float v = x1[idx] + f[idx];
    red[t] = v; __syncthreads();
    for (int s = DIM / 2; s > 0; s >>= 1) { if (t < s) red[t] += red[t + s]; __syncthreads(); }
    float mu = red[0] * (1.f / DIM);
    __syncthreads();
    float d = v - mu;
    red[t] = d * d; __syncthreads();
    for (int s = DIM / 2; s > 0; s >>= 1) { if (t < s) red[t] += red[t + s]; __syncthreads(); }
    out[idx] = d * rsqrtf(red[0] * (1.f / DIM) + LN_EPS) * lw[t] + lb[t];
}

// ---------------------------------------------------------------------------
// launch
// ---------------------------------------------------------------------------
extern "C" void kernel_launch(void* const* d_in, const int* in_sizes, int n_in,
                              void* d_out, int out_size, void* d_ws, size_t ws_size,
                              hipStream_t stream) {
    const float*     x        = (const float*)d_in[0];
    const long long* ei       = (const long long*)d_in[1];   // jnp.int64 edge_index [2,E]
    const float*     W_gat    = (const float*)d_in[2];
    const float*     att_src  = (const float*)d_in[3];
    const float*     att_dst  = (const float*)d_in[4];
    const float*     bias_gat = (const float*)d_in[5];
    const float*     ln_w     = (const float*)d_in[6];
    const float*     ln_b     = (const float*)d_in[7];
    const float*     w1       = (const float*)d_in[8];
    const float*     b1       = (const float*)d_in[9];
    const float*     w2       = (const float*)d_in[10];
    const float*     b2       = (const float*)d_in[11];
    float*           out      = (float*)d_out;

    const int Nn = in_sizes[0] / DIM;     // 50000
    const int E_ = in_sizes[1] / 2;       // 800000
    const int Et = E_ + Nn;

    const int TB = 256;
    const int gemm_blocks = (Nn + 127) / 128;            // 8 waves x 16 rows each
    const int Npad = gemm_blocks * 128;                  // padded rows: no GEMM bounds checks

    // workspace layout; GEMM-touched buffers are Npad rows (padding rows are
    // written-but-never-read or read-but-never-affect real rows)
    float* h     = (float*)d_ws;                         // Npad*128 f32 (GEMM out)
    float* a_s   = h     + (size_t)Npad * DIM;           // Nn*4
    float* a_d   = a_s   + (size_t)Nn * HEADS;
    float* m     = a_d   + (size_t)Nn * HEADS;
    float* z     = m     + (size_t)Nn * HEADS;
    float* accum = z     + (size_t)Nn * HEADS;           // Nn*128
    float* x1    = accum + (size_t)Nn * DIM;             // Nn*128
    unsigned short* x1b = (unsigned short*)(x1 + (size_t)Nn * DIM);  // Npad*128 bf16
    unsigned short* y   = x1b + (size_t)Npad * DIM;      // Npad*256 bf16
    unsigned short* xb  = y   + (size_t)Npad * 2 * DIM;  // Npad*128 bf16
    unsigned short* Wgp = xb  + (size_t)Npad * DIM;      //  8*4*32*16 = 16384
    unsigned short* w1p = Wgp + 8 * 4 * 32 * 16;         // 16*4*32*16 = 32768
    unsigned short* w2p = w1p + 16 * 4 * 32 * 16;        //  8*8*32*16 = 32768
    float* f     = h;    // reuse: h is dead after edge_scatter_kernel

    init_kernel<<<(Nn * DIM + TB - 1) / TB, TB, 0, stream>>>(m, z, accum, Nn);
    cvt_bf16_kernel<<<(Nn * DIM + TB - 1) / TB, TB, 0, stream>>>(x, xb, Nn * DIM);
    pack_b_kernel<<<(8 * 4 * 32 + TB - 1) / TB, TB, 0, stream>>>(W_gat, DIM, 8, 4, Wgp);
    pack_b_kernel<<<(16 * 4 * 32 + TB - 1) / TB, TB, 0, stream>>>(w1, 2 * DIM, 16, 4, w1p);
    pack_b_kernel<<<(8 * 8 * 32 + TB - 1) / TB, TB, 0, stream>>>(w2, DIM, 8, 8, w2p);

    gemm_h_kernel<<<gemm_blocks, TB, 0, stream>>>(xb, Wgp, h);
    attn_coef_kernel<<<(Nn * HEADS + TB - 1) / TB, TB, 0, stream>>>(h, att_src, att_dst,
                                                                    a_s, a_d, Nn);
    edge_max_kernel<<<(Et + TB - 1) / TB, TB, 0, stream>>>(ei, a_s, a_d, m, E_, Nn);
    edge_sum_kernel<<<(Et + TB - 1) / TB, TB, 0, stream>>>(ei, a_s, a_d, m, z, E_, Nn);
    edge_scatter_kernel<<<(Et * HEADS + TB - 1) / TB, TB, 0, stream>>>(ei, a_s, a_d, m, z,
                                                                       h, accum, E_, Nn);
    ln1_kernel<<<Nn, DIM, 0, stream>>>(x, accum, bias_gat, ln_w, ln_b, x1, x1b);
    gemm_ffn1_kernel<<<gemm_blocks, TB, 0, stream>>>(x1b, w1p, b1, y);
    gemm_ffn2_kernel<<<gemm_blocks, TB, 0, stream>>>(y, w2p, b2, f);
    ln2_kernel<<<Nn, DIM, 0, stream>>>(x1, f, ln_w, ln_b, out);
}
